// RefNonLocal_60644938219913
// MI455X (gfx1250) — matrix operand order repaired and checked by reading, compiled, not run on hardware
//
#include <hip/hip_runtime.h>
#include <hip/hip_bf16.h>
#include <stdint.h>

// ---------------------------------------------------------------------------
// Non-local block: theta = x@Wt; phi = pool(x@Wp); g = pool(x@Wg)
//                  y = softmax(theta phi^T) g ; out = x + y
// B=8, N=4096, M=2048, C=256.  All GEMMs via V_WMMA_F32_16X16X32_BF16.
// phi tiles streamed with GLOBAL_LOAD_ASYNC_TO_LDS_B128 (ASYNCcnt).
// ---------------------------------------------------------------------------

#define BATCH 8
#define NROW  4096
#define MROW  2048
#define CDIM  256

typedef __attribute__((ext_vector_type(16))) __bf16 v16bf;
typedef __attribute__((ext_vector_type(8)))  __bf16 v8bf;
typedef __attribute__((ext_vector_type(8)))  float  v8f;

__device__ __forceinline__ __bf16 f2bf(float f) { return (__bf16)f; }  // hw cvt

// Load a 16-element bf16 fragment as two 8-element groups: p[0..7], p[gap..gap+7].
// gap=8  -> 16 contiguous K values per lane (unused; B-fragments also use gap=16)
// gap=16 -> A-fragment (K groups +0..7 and +16..23 per ISA 16-bit A layout)
__device__ __forceinline__ v16bf ld16(const __bf16* p, int gap) {
    v8bf lo = *(const v8bf*)p;
    v8bf hi = *(const v8bf*)(p + gap);
    v16bf r;
#pragma unroll
    for (int i = 0; i < 8; i++) { r[i] = lo[i]; r[i + 8] = hi[i]; }
    return r;
}

__device__ __forceinline__ v8f wmma_bf16(v16bf a, v16bf b, v8f c) {
    return __builtin_amdgcn_wmma_f32_16x16x32_bf16(
        /*neg_a=*/false, a, /*neg_b=*/false, b,
        /*c_mod=*/(short)0, c, /*reuse_a=*/false, /*reuse_b=*/false);
}

// Async 16B global -> LDS copy (GLOBAL_LOAD_ASYNC_TO_LDS_B128, ASYNCcnt).
// ldsoff: byte offset into workgroup LDS (low 32 bits of generic pointer).
__device__ __forceinline__ void async_g2l_b128(uint32_t ldsoff, const void* gaddr) {
    asm volatile("global_load_async_to_lds_b128 %0, %1, off"
                 :: "v"(ldsoff), "v"(gaddr) : "memory");
}
__device__ __forceinline__ void wait_async0() {
    asm volatile("s_wait_asynccnt 0" ::: "memory");
}

// ---------------------------------------------------------------------------
// Kernel 0: weight transpose + bf16 convert.  wt[n][k] = w[k][n]
// ---------------------------------------------------------------------------
__global__ __launch_bounds__(256)
void wt_transpose_kernel(const float* __restrict__ w0,
                         const float* __restrict__ w1,
                         const float* __restrict__ w2,
                         __bf16* __restrict__ wt) {
    int bid  = blockIdx.x;          // 3*256 blocks
    int wsel = bid >> 8;
    int n    = bid & 255;
    const float* w = (wsel == 0) ? w0 : (wsel == 1) ? w1 : w2;
    int k = threadIdx.x;
    wt[(size_t)wsel * CDIM * CDIM + (size_t)n * CDIM + k] = f2bf(w[(size_t)k * CDIM + n]);
}

// ---------------------------------------------------------------------------
// Kernel 1: fused projections + maxpool.
// Block = 256 threads (8 waves), 32 consecutive rows of x (one batch).
// ---------------------------------------------------------------------------
#define LXP 280   // LDS row stride (elements): 560B -> conflict-free A-frag reads
__global__ __launch_bounds__(256)
void proj_pool_kernel(const float* __restrict__ x,
                      const __bf16* __restrict__ wt,
                      __bf16* __restrict__ th,
                      __bf16* __restrict__ ph,
                      __bf16* __restrict__ gg) {
    __shared__ __bf16 lx[32][LXP];

    int b    = blockIdx.x >> 7;              // / 128
    int rb   = blockIdx.x & 127;
    int row0 = b * NROW + rb * 32;
    int t    = threadIdx.x;

#pragma unroll 4
    for (int j = 0; j < 32; j++)
        lx[j][t] = f2bf(x[(size_t)(row0 + j) * CDIM + t]);
    __syncthreads();

    int wave = t >> 5, lane = t & 31, half = lane >> 4, ln = lane & 15;
    int prow0 = b * MROW + rb * 16;

    for (int wsel = 0; wsel < 3; wsel++) {
        const __bf16* W = wt + (size_t)wsel * CDIM * CDIM;
#pragma unroll
        for (int c2 = 0; c2 < 2; c2++) {
            int ct   = wave * 2 + c2;        // 16 col tiles over 8 waves
            int ncol = ct * 16 + ln;
            v16bf bf[8];
#pragma unroll
            for (int kc = 0; kc < 8; kc++)
                bf[kc] = ld16(W + (size_t)ncol * CDIM + kc * 32 + half * 8, 16);

            v8f acc[2];
#pragma unroll
            for (int rt = 0; rt < 2; rt++) {
                v8f c = {};
#pragma unroll
                for (int kc = 0; kc < 8; kc++) {
                    v16bf a = ld16(&lx[rt * 16 + ln][kc * 32 + half * 8], 16);
                    c = wmma_bf16(a, bf[kc], c);
                }
                acc[rt] = c;
            }

            if (wsel == 0) {
#pragma unroll
                for (int rt = 0; rt < 2; rt++)
#pragma unroll
                    for (int r = 0; r < 8; r++) {
                        int m = rt * 16 + r + half * 8;
                        th[(size_t)(row0 + m) * CDIM + ncol] = f2bf(acc[rt][r]);
                    }
            } else {
                __bf16* dst = (wsel == 1) ? ph : gg;
#pragma unroll
                for (int rt = 0; rt < 2; rt++)
#pragma unroll
                    for (int i = 0; i < 4; i++) {
                        float v = fmaxf(acc[rt][2 * i], acc[rt][2 * i + 1]);
                        int p = rt * 8 + half * 4 + i;       // pooled row in [0,16)
                        dst[(size_t)(prow0 + p) * CDIM + ncol] = f2bf(v);
                    }
            }
        }
    }
}

// ---------------------------------------------------------------------------
// Kernel 2: flash attention.  Block = 256 threads (8 waves), 128 query rows.
// theta tiles live in per-wave LDS (A-fragments re-read per K-chunk);
// phi tiles double-buffered via async global->LDS; g^T via scalar transpose.
// ---------------------------------------------------------------------------
#define PHP 264   // phi/theta LDS row stride: 528B, 16B aligned, conflict-free
#define GTP 40    // g^T / P LDS row stride: 80B, 16B aligned, conflict-free
__global__ __launch_bounds__(256)
void attn_kernel(const float* __restrict__ x,
                 const __bf16* __restrict__ th,
                 const __bf16* __restrict__ ph,
                 const __bf16* __restrict__ gg,
                 float* __restrict__ out) {
    __shared__ __bf16 sphi[2][32][PHP];      // keys row-major (B-frags for QK^T)
    __shared__ __bf16 sgt[2][CDIM][GTP];     // g transposed  (B-frags for PV)
    __shared__ __bf16 spb[8][16][GTP];       // per-wave P scratch (C->A relayout)
    __shared__ __bf16 sth[8][16][PHP];       // per-wave theta (A-frags)

    int t = threadIdx.x, wave = t >> 5, lane = t & 31, half = lane >> 4, ln = lane & 15;
    int b     = blockIdx.x >> 5;
    int qb    = blockIdx.x & 31;
    int qrow0 = b * NROW + qb * 128 + wave * 16;

    // stage this wave's 16x256 theta tile into LDS (16B chunks, coalesced)
    const __bf16* thw = th + (size_t)qrow0 * CDIM;
    {
#pragma unroll
        for (int j = 0; j < 16; j++) {
            int c = lane + 32 * j;           // 512 16B-chunks, 16 per lane
            int r = c >> 5, co = (c & 31) * 8;
            *(v8bf*)&sth[wave][r][co] = *(const v8bf*)(thw + (size_t)r * CDIM + co);
        }
    }

    v8f yacc[16] = {};
    float rmax[8], rsum[8];
#pragma unroll
    for (int r = 0; r < 8; r++) { rmax[r] = -3.0e38f; rsum[r] = 0.0f; }

    const __bf16* phb = ph + (size_t)b * MROW * CDIM;
    const __bf16* ggb = gg + (size_t)b * MROW * CDIM;

    auto fill = [&](int buf, int chunk) {
        int kb = chunk * 32;
        // phi tile: 1024 16B chunks via async global->LDS (no VGPR staging)
#pragma unroll
        for (int j = 0; j < 4; j++) {
            int c = j * 256 + t;             // chunk id; 32 chunks per 256-el row
            int r = c >> 5, co = (c & 31) * 8;
            uint32_t ldsoff = (uint32_t)(uintptr_t)&sphi[buf][r][co];
            async_g2l_b128(ldsoff, phb + (size_t)(kb + r) * CDIM + co);
        }
        // g tile transposed: coalesced reads, scattered 2B LDS writes
#pragma unroll 4
        for (int key = 0; key < 32; key++)
            sgt[buf][t][key] = ggb[(size_t)(kb + key) * CDIM + t];
    };

    fill(0, 0);
    wait_async0();
    __syncthreads();

    const float L2E = 1.4426950408889634f;
    for (int chunk = 0; chunk < MROW / 32; chunk++) {
        int buf = chunk & 1;
        if (chunk + 1 < MROW / 32) fill(buf ^ 1, chunk + 1);

        // ---- S = theta * phi^T  (two 16x16 key tiles) ----
        v8f s0 = {}, s1 = {};
#pragma unroll
        for (int kc = 0; kc < 8; kc++) {
            v16bf a  = ld16(&sth[wave][ln][kc * 32 + half * 8], 16);
            v16bf b0 = ld16(&sphi[buf][ln][kc * 32 + half * 8], 16);
            v16bf b1 = ld16(&sphi[buf][16 + ln][kc * 32 + half * 8], 16);
            s0 = wmma_bf16(a, b0, s0);
            s1 = wmma_bf16(a, b1, s1);
        }

        // ---- online softmax (exp2 domain, 16-lane butterflies) ----
        float corr[8];
#pragma unroll
        for (int r = 0; r < 8; r++) {
            float a0 = s0[r] * L2E, a1 = s1[r] * L2E;
            float m = fmaxf(a0, a1);
            m = fmaxf(m, __shfl_xor(m, 1));
            m = fmaxf(m, __shfl_xor(m, 2));
            m = fmaxf(m, __shfl_xor(m, 4));
            m = fmaxf(m, __shfl_xor(m, 8));
            float nm = fmaxf(rmax[r], m);
            float c  = exp2f(rmax[r] - nm);
            float p0 = exp2f(a0 - nm);
            float p1 = exp2f(a1 - nm);
            float ps = p0 + p1;
            ps += __shfl_xor(ps, 1);
            ps += __shfl_xor(ps, 2);
            ps += __shfl_xor(ps, 4);
            ps += __shfl_xor(ps, 8);
            rsum[r] = rsum[r] * c + ps;
            rmax[r] = nm;
            corr[r] = c;
            int ml = r + half * 8;           // C-layout row
            spb[wave][ml][ln]      = f2bf(p0);
            spb[wave][ml][16 + ln] = f2bf(p1);
        }
#pragma unroll
        for (int ctt = 0; ctt < 16; ctt++)
#pragma unroll
            for (int r = 0; r < 8; r++) yacc[ctt][r] *= corr[r];

        // ---- y += P * g  (P as A-fragment, K = 32 keys) ----
        v16bf pf = ld16(&spb[wave][ln][half * 8], 16);
#pragma unroll
        for (int ctt = 0; ctt < 16; ctt++) {
            v16bf bg = ld16(&sgt[buf][ctt * 16 + ln][half * 8], 16);
            yacc[ctt] = wmma_bf16(pf, bg, yacc[ctt]);
        }

        wait_async0();      // next phi tile landed in LDS
        __syncthreads();    // + all waves done with current tiles (ds drained)
    }

    // ---- epilogue: out = x + y / rowsum ----
    float inv[8];
#pragma unroll
    for (int r = 0; r < 8; r++) inv[r] = 1.0f / rsum[r];
    const float* xb = x + (size_t)qrow0 * CDIM;
    float* ob = out + (size_t)qrow0 * CDIM;
#pragma unroll
    for (int ctt = 0; ctt < 16; ctt++)
#pragma unroll
        for (int r = 0; r < 8; r++) {
            size_t idx = (size_t)(r + half * 8) * CDIM + ctt * 16 + ln;
            ob[idx] = xb[idx] + yacc[ctt][r] * inv[r];
        }
}

// ---------------------------------------------------------------------------
extern "C" void kernel_launch(void* const* d_in, const int* in_sizes, int n_in,
                              void* d_out, int out_size, void* d_ws, size_t ws_size,
                              hipStream_t stream) {
    (void)in_sizes; (void)n_in; (void)out_size; (void)ws_size;
    const float* x  = (const float*)d_in[0];
    const float* w0 = (const float*)d_in[1];
    const float* w1 = (const float*)d_in[2];
    const float* w2 = (const float*)d_in[3];
    float* out = (float*)d_out;

    // workspace layout (bf16):  wt[3*C*C] | theta[B*N*C] | phi[B*M*C] | g[B*M*C]
    char* wsb = (char*)d_ws;
    __bf16* wt = (__bf16*)wsb;
    __bf16* th = (__bf16*)(wsb + (size_t)3 * CDIM * CDIM * 2);
    __bf16* ph = th + (size_t)BATCH * NROW * CDIM;
    __bf16* gg = ph + (size_t)BATCH * MROW * CDIM;

    wt_transpose_kernel<<<3 * 256, 256, 0, stream>>>(w0, w1, w2, wt);
    proj_pool_kernel<<<BATCH * (NROW / 32), 256, 0, stream>>>(x, wt, th, ph, gg);
    attn_kernel<<<BATCH * (NROW / 128), 256, 0, stream>>>(x, th, ph, gg, out);
}
